// LanePointNetEncoder_26371099197706
// MI455X (gfx1250) — matrix operand order repaired and checked by reading, not compiled, not run on hardware
//
#include <hip/hip_runtime.h>
#include <hip/hip_bf16.h>

// ---------------------------------------------------------------------------
// Types for CDNA5 WMMA (gfx1250, wave32)
// ---------------------------------------------------------------------------
typedef __attribute__((ext_vector_type(16))) __bf16 v16bf;
typedef __attribute__((ext_vector_type(8)))  float  v8f;

union V16U { uint4 q[2]; v16bf v; };

__device__ __forceinline__ unsigned short f2bf(float f) {
    unsigned int u = __float_as_uint(f);
    unsigned int r = (u + 0x7FFFu + ((u >> 16) & 1u)) >> 16;  // RNE
    return (unsigned short)r;
}
__device__ __forceinline__ float bf2f(unsigned short h) {
    return __uint_as_float(((unsigned int)h) << 16);
}

// ---------------------------------------------------------------------------
// Prep: convert f32 weight matrix (cin x 128, row-major) into bf16 WMMA
// B-operand fragments, K zero-padded to multiples of 32.
// Layout: frag[((kc*2 + khalf)*128 + n)*16 + j] = W[(kc*32 + khalf*16 + j), n]
// -> each lane's B fragment is one contiguous 32B read (coalesced).
// ---------------------------------------------------------------------------
__global__ void build_wfrag(const float* __restrict__ W,
                            unsigned short* __restrict__ dst, int cin) {
    int idx = blockIdx.x * blockDim.x + threadIdx.x;
    int total = ((cin + 31) / 32) * 2 * 128 * 16;
    if (idx >= total) return;
    int j  = idx & 15;
    int n  = (idx >> 4) & 127;
    int kh = (idx >> 11) & 1;
    int kc = idx >> 12;
    int k  = kc * 32 + kh * 16 + j;
    float v = (k < cin) ? W[(size_t)k * 128 + n] : 0.0f;
    dst[idx] = f2bf(v);
}

// Fragment offsets (in halfs) inside d_ws
#define OFF_PRE0 0          // K=32 (5 padded):  2*128*16      = 4096
#define OFF_PRE1 4096       // K=128:            4*2*128*16    = 16384
#define OFF_PRE2 20480      // K=128
#define OFF_MID0 36864      // K=256:            8*2*128*16    = 32768
#define OFF_MID1 69632      // K=128
// total 86016 halfs = 172032 bytes of d_ws

// ---------------------------------------------------------------------------
// Main kernel: one block per (b,m) lane. 256 threads = 8 waves.
// Wave w computes output columns [16w, 16w+16) over all 64 rows
// (4 row-tiles of 16x16, f32 accumulators).
// ---------------------------------------------------------------------------
__global__ __launch_bounds__(256) void lane_enc(
    const float* __restrict__ pos, const float* __restrict__ attr,
    const int*  __restrict__ pmask, const int* __restrict__ kmask,
    const unsigned short* __restrict__ wfrag,
    const float* __restrict__ s0, const float* __restrict__ t0,
    const float* __restrict__ s1, const float* __restrict__ t1,
    const float* __restrict__ s2, const float* __restrict__ t2,
    const float* __restrict__ sm0, const float* __restrict__ tm0,
    const float* __restrict__ sm1, const float* __restrict__ tm1,
    const float* __restrict__ W0, const float* __restrict__ b0,
    const float* __restrict__ W1, const float* __restrict__ b1,
    float* __restrict__ out)
{
    __shared__ __align__(16) unsigned short act[64 * 256];  // bf16 activations
    __shared__ float spool[128];
    __shared__ float smask[64];
    __shared__ float sy[128];

    const int bm     = blockIdx.x;      // 0..4095  == b*128 + m
    const int tid    = threadIdx.x;
    const int lane   = tid & 31;
    const int wave   = tid >> 5;
    const int lanelo = lane & 15;
    const int khalf  = lane >> 4;
    const int n0     = wave * 16;       // column strip of this wave

    // ---- stage 0: build [64,5] input features into act cols 0..31 ----
    if (tid < 64) {
        int l = tid;
        const float* p = pos + ((size_t)bm * 64 + l) * 2;
        float vx = 0.0f, vy = 0.0f;
        if (l > 0) { vx = p[0] - p[-2]; vy = p[1] - p[-1]; }
        float ang = atan2f(vy, vx);
        float lt  = attr[bm * 3 + 0];
        float lw  = attr[bm * 3 + 2];
        unsigned short* row = act + l * 256;
        row[0] = f2bf(vx); row[1] = f2bf(vy); row[2] = f2bf(ang);
        row[3] = f2bf(lt); row[4] = f2bf(lw);
        #pragma unroll
        for (int c = 5; c < 32; ++c) row[c] = 0;
        smask[l] = pmask[bm * 64 + l] ? 0.0f : 1.0f;
    }
    __syncthreads();

    // ---- GEMM over k-chunks of 32, accumulating 4 row-tiles ----
    auto gemm = [&](const unsigned short* __restrict__ wf, int kchunks, v8f acc[4]) {
        for (int kc = 0; kc < kchunks; ++kc) {
            V16U b;
            const uint4* bp = (const uint4*)(wf +
                (((size_t)(kc * 2 + khalf) * 128 + n0 + lanelo) << 4));
            b.q[0] = bp[0];
            b.q[1] = bp[1];
            #pragma unroll
            for (int rt = 0; rt < 4; ++rt) {
                V16U a;
                const unsigned short* ap =
                    act + (rt * 16 + lanelo) * 256 + kc * 32 + khalf * 16;
                a.q[0] = *(const uint4*)(ap);
                a.q[1] = *(const uint4*)(ap + 8);
                acc[rt] = __builtin_amdgcn_wmma_f32_16x16x32_bf16(
                    false, a.v, false, b.v, (short)0, acc[rt], false, false);
            }
        }
    };

    // ---- BN affine + ReLU (+ optional row mask), bf16 writeback to LDS ----
    auto writeback = [&](v8f acc[4], const float* s, const float* t, bool maskrows) {
        float sc = s[n0 + lanelo], sh = t[n0 + lanelo];
        __syncthreads();   // all waves done reading act
        #pragma unroll
        for (int rt = 0; rt < 4; ++rt) {
            #pragma unroll
            for (int r = 0; r < 8; ++r) {
                int row = rt * 16 + khalf * 8 + r;
                float v = fmaxf(acc[rt][r] * sc + sh, 0.0f);
                if (maskrows) v *= smask[row];
                act[row * 256 + n0 + lanelo] = f2bf(v);
            }
        }
        __syncthreads();
    };

    v8f acc[4];
    const v8f vzero = {};

    // pre-MLP layer 0: K=32 (5 padded)
    #pragma unroll
    for (int rt = 0; rt < 4; ++rt) acc[rt] = vzero;
    gemm(wfrag + OFF_PRE0, 1, acc);
    writeback(acc, s0, t0, false);

    // pre-MLP layer 1: K=128
    #pragma unroll
    for (int rt = 0; rt < 4; ++rt) acc[rt] = vzero;
    gemm(wfrag + OFF_PRE1, 4, acc);
    writeback(acc, s1, t1, false);

    // pre-MLP layer 2: K=128, masked output
    #pragma unroll
    for (int rt = 0; rt < 4; ++rt) acc[rt] = vzero;
    gemm(wfrag + OFF_PRE2, 4, acc);
    writeback(acc, s2, t2, true);

    // max-pool over L, broadcast into concat cols 128..255
    if (tid < 128) {
        float mm = bf2f(act[tid]);
        for (int r = 1; r < 64; ++r) mm = fmaxf(mm, bf2f(act[r * 256 + tid]));
        spool[tid] = mm;
    }
    __syncthreads();
    for (int idx = tid; idx < 64 * 128; idx += 256) {
        int r = idx >> 7, c = idx & 127;
        act[r * 256 + 128 + c] = f2bf(spool[c]);
    }
    __syncthreads();

    // mid layer 0: K=256
    #pragma unroll
    for (int rt = 0; rt < 4; ++rt) acc[rt] = vzero;
    gemm(wfrag + OFF_MID0, 8, acc);
    writeback(acc, sm0, tm0, false);

    // mid layer 1: K=128, masked output
    #pragma unroll
    for (int rt = 0; rt < 4; ++rt) acc[rt] = vzero;
    gemm(wfrag + OFF_MID1, 4, acc);
    writeback(acc, sm1, tm1, true);

    // max-pool over L -> pooled2 (f32)
    if (tid < 128) {
        float mm = bf2f(act[tid]);
        for (int r = 1; r < 64; ++r) mm = fmaxf(mm, bf2f(act[r * 256 + tid]));
        spool[tid] = mm;
    }
    __syncthreads();

    // out MLP (f32 exact): relu(pooled2 @ W0 + b0) @ W1 + b1, lane mask
    if (tid < 128) {
        float a = b0[tid];
        for (int k = 0; k < 128; ++k) a += spool[k] * W0[(size_t)k * 128 + tid];
        sy[tid] = fmaxf(a, 0.0f);
    }
    __syncthreads();
    if (tid < 128) {
        float a = b1[tid];
        for (int k = 0; k < 128; ++k) a += sy[k] * W1[(size_t)k * 128 + tid];
        float valid = kmask[bm] ? 0.0f : 1.0f;
        out[(size_t)bm * 128 + tid] = a * valid;
    }
}

// ---------------------------------------------------------------------------
// Launch
// ---------------------------------------------------------------------------
extern "C" void kernel_launch(void* const* d_in, const int* in_sizes, int n_in,
                              void* d_out, int out_size, void* d_ws, size_t ws_size,
                              hipStream_t stream) {
    (void)in_sizes; (void)n_in; (void)out_size; (void)ws_size;
    const float* pos   = (const float*)d_in[0];
    const float* attr  = (const float*)d_in[1];
    const int*   pmask = (const int*)d_in[2];
    const int*   kmask = (const int*)d_in[3];
    const float* preW0 = (const float*)d_in[4];
    const float* preS0 = (const float*)d_in[5];
    const float* preT0 = (const float*)d_in[6];
    const float* preW1 = (const float*)d_in[7];
    const float* preS1 = (const float*)d_in[8];
    const float* preT1 = (const float*)d_in[9];
    const float* preW2 = (const float*)d_in[10];
    const float* preS2 = (const float*)d_in[11];
    const float* preT2 = (const float*)d_in[12];
    const float* midW0 = (const float*)d_in[13];
    const float* midS0 = (const float*)d_in[14];
    const float* midT0 = (const float*)d_in[15];
    const float* midW1 = (const float*)d_in[16];
    const float* midS1 = (const float*)d_in[17];
    const float* midT1 = (const float*)d_in[18];
    const float* outW0 = (const float*)d_in[19];
    const float* outB0 = (const float*)d_in[20];
    const float* outW1 = (const float*)d_in[21];
    const float* outB1 = (const float*)d_in[22];
    float* out = (float*)d_out;

    unsigned short* wfrag = (unsigned short*)d_ws;

    // Convert weights to bf16 WMMA B-fragments (deterministic every call).
    build_wfrag<<<16,  256, 0, stream>>>(preW0, wfrag + OFF_PRE0, 5);
    build_wfrag<<<64,  256, 0, stream>>>(preW1, wfrag + OFF_PRE1, 128);
    build_wfrag<<<64,  256, 0, stream>>>(preW2, wfrag + OFF_PRE2, 128);
    build_wfrag<<<128, 256, 0, stream>>>(midW0, wfrag + OFF_MID0, 256);
    build_wfrag<<<64,  256, 0, stream>>>(midW1, wfrag + OFF_MID1, 128);

    // One block per (b,m) lane: 32*128 = 4096 blocks, 8 waves each.
    lane_enc<<<4096, 256, 0, stream>>>(
        pos, attr, pmask, kmask, wfrag,
        preS0, preT0, preS1, preT1, preS2, preT2,
        midS0, midT0, midS1, midT1,
        outW0, outB0, outW1, outB1, out);
}
